// AudioImaginationForGLUE_24361054503184
// MI455X (gfx1250) — compile-verified, matrix operand order checked by reading, not run on hardware
//
#include <hip/hip_runtime.h>
#include <hip/hip_bf16.h>
#include <math.h>

// ---------------- problem constants ----------------
constexpr int Bb = 32, Ss = 512, Hh = 768, NHh = 12, FFf = 3072;
constexpr int Aa = 768, TAa = 1024, NSPANn = 2, MAXLl = 64, DHh = 64;
#define ATT_SCALE 0.125f  // 1/sqrt(64)

typedef _Float16 h16;
typedef _Float16 v16h __attribute__((ext_vector_type(16)));
typedef _Float16 v8h  __attribute__((ext_vector_type(8)));
typedef float    v8f  __attribute__((ext_vector_type(8)));
typedef unsigned int u32x4 __attribute__((ext_vector_type(4)));
typedef int i32x4 __attribute__((ext_vector_type(4)));
typedef int i32x8 __attribute__((ext_vector_type(8)));

union Frag16 { v16h v; v8h p[2]; };

// Load a 16x32 f16 WMMA A/B fragment from a row-major (rows x K) matrix.
// ISA layout (CDNA5 ISA): lanes 0-15 row=lane, K in {0..7,16..23};
// lanes 16-31 row=lane-16, K in {8..15,24..31}. Two contiguous 16B loads/lane.
__device__ __forceinline__ v16h load_ab(const h16* __restrict__ p, int ld) {
  const int lane = threadIdx.x & 31;
  const int r = lane & 15;
  const int ko = (lane & 16) ? 8 : 0;
  Frag16 f;
  f.p[0] = *(const v8h*)(p + (size_t)r * ld + ko);
  f.p[1] = *(const v8h*)(p + (size_t)r * ld + ko + 16);
  return f.v;
}

__device__ __forceinline__ v8f wmma16(v16h a, v16h b, v8f c) {
  return __builtin_amdgcn_wmma_f32_16x16x32_f16(false, a, false, b, (short)0, c,
                                                false, false);
}

// ---------------- generic WMMA GEMM ----------------
// C(M,N) = act(A(M,K) @ Bt(N,K)^T + bias).
// Block = 4 waves (2x2), block tile 64x128, wave tile 32x64:
// per k-step 2 A frags + 4 B frags -> 8 WMMAs (2x the FLOP/byte of a 16-row
// wave tile; the 32768x768x768 GEMMs dominate so L2 traffic/FLOP matters).
// ACT: 0 none, 1 relu, 2 exact gelu.
template <int ACT, bool WH, bool WF>
__global__ __launch_bounds__(128) void gemm_kernel(
    const h16* __restrict__ A, const h16* __restrict__ Bt,
    const float* __restrict__ bias, h16* __restrict__ outh,
    float* __restrict__ outf, int M, int N, int K) {
  const int wave = threadIdx.x >> 5;
  const int lane = threadIdx.x & 31;
  const int tn = blockIdx.x * 128 + (wave >> 1) * 64;
  const int tm = blockIdx.y * 64 + (wave & 1) * 32;
  const h16* Ab = A + (size_t)tm * K;
  v8f acc[2][4] = {};
  for (int k = 0; k < K; k += 32) {
    const v16h a0 = load_ab(Ab + k, K);
    const v16h a1 = load_ab(Ab + (size_t)16 * K + k, K);
    if (k + 64 < K)  // stream-ahead hint on the A stream (global_prefetch_b8)
      __builtin_prefetch(Ab + (size_t)(lane & 15) * K + ((lane & 16) ? 8 : 0) +
                             k + 64, 0, 0);
#pragma unroll
    for (int j = 0; j < 4; ++j) {
      const v16h b = load_ab(Bt + (size_t)(tn + j * 16) * K + k, K);
      acc[0][j] = wmma16(a0, b, acc[0][j]);
      acc[1][j] = wmma16(a1, b, acc[1][j]);
    }
  }
  // C layout: lane&15 = column, VGPR r = row (lanes>=16 -> rows 8..15)
  const int col0 = lane & 15;
  const int rbase = (lane & 16) ? 8 : 0;
#pragma unroll
  for (int i = 0; i < 2; ++i) {
#pragma unroll
    for (int j = 0; j < 4; ++j) {
      const int col = tn + j * 16 + col0;
      const float bv = bias[col];
#pragma unroll
      for (int r = 0; r < 8; ++r) {
        float x = acc[i][j][r] + bv;
        if (ACT == 1) x = fmaxf(x, 0.f);
        if (ACT == 2) x = 0.5f * x * (1.f + erff(x * 0.70710678118f));
        const size_t o = (size_t)(tm + i * 16 + rbase + r) * N + col;
        if (WH) outh[o] = (h16)x;
        if (WF) outf[o] = x;
      }
    }
  }
}

// ---------------- TDM compile-probe (not launched) ----------------
// Validates the Tensor Data Mover encoding path (tensor_load_to_lds +
// s_wait_tensorcnt) with a NULL descriptor (D# group0 count=0), for a future
// LDS-staged GEMM. Arity differs per toolchain (bridge doc): clang-22 = 5 args,
// clang-23 = 6 args.
#if defined(__has_builtin)
#if __has_builtin(__builtin_amdgcn_tensor_load_to_lds)
__global__ void tdm_probe_kernel(int* sink) {
  u32x4 g0 = {};
  i32x8 g1 = {};
  i32x4 g2 = {};
  i32x4 g3 = {};
#if __clang_major__ >= 23
  i32x8 g4 = {};
  __builtin_amdgcn_tensor_load_to_lds(g0, g1, g2, g3, g4, 0);
#else
  __builtin_amdgcn_tensor_load_to_lds(g0, g1, g2, g3, 0);
#endif
  __builtin_amdgcn_s_wait_tensorcnt(0);
  if (sink) *sink = 1;
}
#endif
#endif

// ---------------- helper kernels ----------------
__global__ __launch_bounds__(256) void copyf_kernel(const float* __restrict__ x,
                                                    float* __restrict__ y,
                                                    size_t n) {
  size_t i = (size_t)blockIdx.x * blockDim.x + threadIdx.x;
  if (i < n) y[i] = x[i];
}

// w (K x N) f32 -> wt (N x K) f16
__global__ __launch_bounds__(256) void wtrans_kernel(const float* __restrict__ w,
                                                     h16* __restrict__ wt, int K,
                                                     int N) {
  size_t i = (size_t)blockIdx.x * blockDim.x + threadIdx.x;
  if (i >= (size_t)K * N) return;
  int k = (int)(i % K);
  int n = (int)(i / K);
  wt[i] = (h16)w[(size_t)k * N + n];
}

// audio (B, NSPAN, TA, A) slice span s -> f16 (B*TA, A)
__global__ __launch_bounds__(256) void cvt_audio_kernel(
    const float* __restrict__ audio, h16* __restrict__ out, int s) {
  size_t i = (size_t)blockIdx.x * blockDim.x + threadIdx.x;
  const size_t per = (size_t)TAa * Aa;
  if (i >= (size_t)Bb * per) return;
  size_t b = i / per, rest = i % per;
  out[i] = (h16)audio[(b * NSPANn + s) * per + rest];
}

// vh (b*TA + t, h*64+dh) -> vth[((b*NH+h)*64+dh)*TA + t]
__global__ __launch_bounds__(256) void vtrans_kernel(const h16* __restrict__ vh,
                                                     h16* __restrict__ vth) {
  size_t i = (size_t)blockIdx.x * blockDim.x + threadIdx.x;
  if (i >= (size_t)Bb * NHh * DHh * TAa) return;
  int t = (int)(i % TAa);
  size_t j = i / TAa;
  int dh = (int)(j % DHh);
  j /= DHh;
  int h = (int)(j % NHh);
  int b = (int)(j / NHh);
  vth[i] = vh[((size_t)b * TAa + t) * Hh + h * DHh + dh];
}

__global__ __launch_bounds__(256) void gather_kernel(
    const float* __restrict__ hs, const int* __restrict__ spans, int s,
    float* __restrict__ spanf, h16* __restrict__ spanh) {
  const int b = blockIdx.x / MAXLl, l = blockIdx.x % MAXLl;
  const int st = spans[(b * NSPANn + s) * 2];
  int en = spans[(b * NSPANn + s) * 2 + 1];
  if (en > Ss) en = Ss;
  int L = en - st;
  if (L < 0) L = 0;
  int idx = st + l;
  idx = idx < 0 ? 0 : (idx > Ss - 1 ? Ss - 1 : idx);
  const bool valid = l < L;
  const size_t orow = ((size_t)b * MAXLl + l) * Hh;
  const size_t irow = ((size_t)b * Ss + idx) * Hh;
  for (int c = threadIdx.x; c < Hh; c += blockDim.x) {
    float v = valid ? hs[irow + c] : 0.f;
    spanf[orow + c] = v;
    spanh[orow + c] = (h16)v;
  }
}

// out = g*(x-mean)*rsqrt(var+1e-5)+b  with x = a + r, row length 768
__global__ __launch_bounds__(256) void addln_kernel(
    const float* __restrict__ a, const float* __restrict__ r,
    const float* __restrict__ g, const float* __restrict__ be,
    float* __restrict__ outf, h16* __restrict__ outh) {
  __shared__ float sred[256];
  const int tid = threadIdx.x;
  const size_t base = (size_t)blockIdx.x * Hh;
  float x0 = a[base + tid] + r[base + tid];
  float x1 = a[base + tid + 256] + r[base + tid + 256];
  float x2 = a[base + tid + 512] + r[base + tid + 512];
  sred[tid] = x0 + x1 + x2;
  __syncthreads();
  for (int st = 128; st > 0; st >>= 1) {
    if (tid < st) sred[tid] += sred[tid + st];
    __syncthreads();
  }
  const float mean = sred[0] * (1.f / Hh);
  __syncthreads();
  const float d0 = x0 - mean, d1 = x1 - mean, d2 = x2 - mean;
  sred[tid] = d0 * d0 + d1 * d1 + d2 * d2;
  __syncthreads();
  for (int st = 128; st > 0; st >>= 1) {
    if (tid < st) sred[tid] += sred[tid + st];
    __syncthreads();
  }
  const float rstd = rsqrtf(sred[0] * (1.f / Hh) + 1e-5f);
  float y0 = g[tid] * d0 * rstd + be[tid];
  float y1 = g[tid + 256] * d1 * rstd + be[tid + 256];
  float y2 = g[tid + 512] * d2 * rstd + be[tid + 512];
  outf[base + tid] = y0;
  outf[base + tid + 256] = y1;
  outf[base + tid + 512] = y2;
  outh[base + tid] = (h16)y0;
  outh[base + tid + 256] = (h16)y1;
  outh[base + tid + 512] = (h16)y2;
}

__global__ __launch_bounds__(256) void scatter_kernel(
    const float* __restrict__ g1, const float* __restrict__ g2,
    const float* __restrict__ out2f, const float* __restrict__ spanf,
    const int* __restrict__ spans, const int* __restrict__ act, int s,
    float* __restrict__ hs) {
  const int b = blockIdx.x / MAXLl, l = blockIdx.x % MAXLl;
  if (act[b * NSPANn + s] == 0) return;
  const int st = spans[(b * NSPANn + s) * 2];
  int en = spans[(b * NSPANn + s) * 2 + 1];
  if (en > Ss) en = Ss;
  int L = en - st;
  if (L < 0) L = 0;
  if (l >= L) return;
  int idx = st + l;
  idx = idx < 0 ? 0 : (idx > Ss - 1 ? Ss - 1 : idx);
  const size_t row = ((size_t)b * MAXLl + l) * Hh;
  const size_t orow = ((size_t)b * Ss + idx) * Hh;
  for (int c = threadIdx.x; c < Hh; c += blockDim.x) {
    float gate = 1.f / (1.f + __expf(-(g1[row + c] + g2[row + c])));
    hs[orow + c] += gate * (out2f[row + c] - spanf[row + c]);
  }
}

// ---------------- flash attention: one block per (b,h), 4 waves ----------------
// wave w owns q rows [16w,16w+16); online softmax over TA=1024 keys in chunks
// of 32; P transposed C-layout -> A-layout through per-wave LDS.
__global__ __launch_bounds__(128) void attn_kernel(const h16* __restrict__ qh,
                                                   const h16* __restrict__ kh,
                                                   const h16* __restrict__ vth,
                                                   h16* __restrict__ ctxh) {
  __shared__ __align__(16) h16 plds[4 * 16 * 32];
  const int b = blockIdx.x / NHh, h = blockIdx.x % NHh;
  const int wave = threadIdx.x >> 5, lane = threadIdx.x & 31;
  const int col0 = lane & 15;
  const int rbase = (lane & 16) ? 8 : 0;
  const h16* qb = qh + ((size_t)b * MAXLl + wave * 16) * Hh + h * DHh;
  const h16* kb = kh + (size_t)b * TAa * Hh + h * DHh;
  const h16* vb = vth + (size_t)(b * NHh + h) * DHh * TAa;
  h16* pl = plds + wave * 512;

  const v16h qf0 = load_ab(qb, Hh);
  const v16h qf1 = load_ab(qb + 32, Hh);
  float m[8], lsum[8];
  v8f o[4] = {};
#pragma unroll
  for (int r = 0; r < 8; ++r) { m[r] = -1e30f; lsum[r] = 0.f; }

  for (int t0 = 0; t0 < TAa; t0 += 32) {
    v8f s0 = {}, s1 = {};
    s0 = wmma16(qf0, load_ab(kb + (size_t)t0 * Hh, Hh), s0);
    s0 = wmma16(qf1, load_ab(kb + (size_t)t0 * Hh + 32, Hh), s0);
    s1 = wmma16(qf0, load_ab(kb + (size_t)(t0 + 16) * Hh, Hh), s1);
    s1 = wmma16(qf1, load_ab(kb + (size_t)(t0 + 16) * Hh + 32, Hh), s1);
    float sm[8], p0[8], p1[8], rs[8];
#pragma unroll
    for (int r = 0; r < 8; ++r) {
      s0[r] *= ATT_SCALE;
      s1[r] *= ATT_SCALE;
      sm[r] = fmaxf(s0[r], s1[r]);
    }
#pragma unroll
    for (int off = 1; off < 16; off <<= 1)
#pragma unroll
      for (int r = 0; r < 8; ++r) sm[r] = fmaxf(sm[r], __shfl_xor(sm[r], off, 32));
#pragma unroll
    for (int r = 0; r < 8; ++r) {
      const float mn = fmaxf(m[r], sm[r]);
      const float alpha = __expf(m[r] - mn);
      m[r] = mn;
      p0[r] = __expf(s0[r] - mn);
      p1[r] = __expf(s1[r] - mn);
      rs[r] = p0[r] + p1[r];
      lsum[r] *= alpha;
#pragma unroll
      for (int j = 0; j < 4; ++j) o[j][r] *= alpha;
    }
#pragma unroll
    for (int off = 1; off < 16; off <<= 1)
#pragma unroll
      for (int r = 0; r < 8; ++r) rs[r] += __shfl_xor(rs[r], off, 32);
#pragma unroll
    for (int r = 0; r < 8; ++r) lsum[r] += rs[r];
    // P (16x32) C-layout -> LDS row-major -> A-fragment
#pragma unroll
    for (int r = 0; r < 8; ++r) {
      pl[(rbase + r) * 32 + col0] = (h16)p0[r];
      pl[(rbase + r) * 32 + 16 + col0] = (h16)p1[r];
    }
    const v16h pf = load_ab(pl, 32);
#pragma unroll
    for (int j = 0; j < 4; ++j)
      o[j] = wmma16(pf, load_ab(vb + (size_t)(j * 16) * TAa + t0, TAa), o[j]);
  }
  h16* cb = ctxh + ((size_t)b * MAXLl + wave * 16) * Hh + h * DHh;
#pragma unroll
  for (int j = 0; j < 4; ++j)
#pragma unroll
    for (int r = 0; r < 8; ++r)
      cb[(size_t)(rbase + r) * Hh + j * 16 + col0] = (h16)(o[j][r] / lsum[r]);
}

// ---------------- host orchestration ----------------
extern "C" void kernel_launch(void* const* d_in, const int* in_sizes, int n_in,
                              void* d_out, int out_size, void* d_ws,
                              size_t ws_size, hipStream_t stream) {
  const float* hs_in = (const float*)d_in[0];
  const float* audio = (const float*)d_in[1];
  const int* spans = (const int*)d_in[2];
  const int* act = (const int*)d_in[3];
  const float* mw1 = (const float*)d_in[4];
  const float* mb1 = (const float*)d_in[5];
  const float* mw2 = (const float*)d_in[6];
  const float* mb2 = (const float*)d_in[7];
  const float* wq = (const float*)d_in[8];
  const float* bq = (const float*)d_in[9];
  const float* wk = (const float*)d_in[10];
  const float* bk = (const float*)d_in[11];
  const float* wv = (const float*)d_in[12];
  const float* bv = (const float*)d_in[13];
  const float* wo = (const float*)d_in[14];
  const float* bo = (const float*)d_in[15];
  const float* ln1g = (const float*)d_in[16];
  const float* ln1b = (const float*)d_in[17];
  const float* fw1 = (const float*)d_in[18];
  const float* fb1 = (const float*)d_in[19];
  const float* fw2 = (const float*)d_in[20];
  const float* fb2 = (const float*)d_in[21];
  const float* ln2g = (const float*)d_in[22];
  const float* ln2b = (const float*)d_in[23];
  const float* gaw = (const float*)d_in[24];
  const float* gab = (const float*)d_in[25];
  const float* gtw = (const float*)d_in[26];
  const float* gtb = (const float*)d_in[27];
  float* hs = (float*)d_out;

  // workspace bump allocator (256B aligned)
  char* wp = (char*)d_ws;
  auto alloc = [&](size_t bytes) -> void* {
    void* r = (void*)wp;
    wp += (bytes + 255) & ~(size_t)255;
    return r;
  };
  const size_t HH2 = (size_t)Hh * Hh * sizeof(h16);
  const size_t HF2 = (size_t)Hh * FFf * sizeof(h16);
  h16* mw1t = (h16*)alloc(HH2);
  h16* mw2t = (h16*)alloc(HH2);
  h16* wqt = (h16*)alloc(HH2);
  h16* wkt = (h16*)alloc(HH2);
  h16* wvt = (h16*)alloc(HH2);
  h16* wot = (h16*)alloc(HH2);
  h16* fw1t = (h16*)alloc(HF2);
  h16* fw2t = (h16*)alloc(HF2);
  h16* gawt = (h16*)alloc(HH2);
  h16* gtwt = (h16*)alloc(HH2);
  const size_t BIG = (size_t)Bb * TAa * Hh * sizeof(h16);  // 50 MB
  h16* bufA = (h16*)alloc(BIG);  // aih, then kh
  h16* bufB = (h16*)alloc(BIG);  // h1h, then vh
  h16* bufC = (h16*)alloc(BIG);  // afh, then vth
  const int ML = Bb * MAXLl;  // 2048
  const int MA = Bb * TAa;    // 32768
  float* spanf = (float*)alloc((size_t)ML * Hh * 4);
  h16* spanh = (h16*)alloc((size_t)ML * Hh * 2);
  h16* qhb = (h16*)alloc((size_t)ML * Hh * 2);
  h16* ctxh = (h16*)alloc((size_t)ML * Hh * 2);
  float* of = (float*)alloc((size_t)ML * Hh * 4);
  float* out1f = (float*)alloc((size_t)ML * Hh * 4);
  h16* out1h = (h16*)alloc((size_t)ML * Hh * 2);
  h16* ffh = (h16*)alloc((size_t)ML * FFf * 2);
  float* ff2f = (float*)alloc((size_t)ML * Hh * 4);
  float* out2f = (float*)alloc((size_t)ML * Hh * 4);
  h16* out2h = (h16*)alloc((size_t)ML * Hh * 2);
  float* g1f = (float*)alloc((size_t)ML * Hh * 4);
  float* g2f = (float*)alloc((size_t)ML * Hh * 4);

  const dim3 blk128(128), blk256(256);
  auto gblocks = [](size_t n) { return dim3((unsigned)((n + 255) / 256)); };
  auto ggemm = [](int M, int N) { return dim3(N / 128, M / 64); };

  // init output = hidden_states
  const size_t nhs = (size_t)Bb * Ss * Hh;
  copyf_kernel<<<gblocks(nhs), blk256, 0, stream>>>(hs_in, hs, nhs);
  // weight transposes f32(KxN) -> f16(NxK)
  wtrans_kernel<<<gblocks((size_t)Aa * Hh), blk256, 0, stream>>>(mw1, mw1t, Aa, Hh);
  wtrans_kernel<<<gblocks((size_t)Hh * Hh), blk256, 0, stream>>>(mw2, mw2t, Hh, Hh);
  wtrans_kernel<<<gblocks((size_t)Hh * Hh), blk256, 0, stream>>>(wq, wqt, Hh, Hh);
  wtrans_kernel<<<gblocks((size_t)Hh * Hh), blk256, 0, stream>>>(wk, wkt, Hh, Hh);
  wtrans_kernel<<<gblocks((size_t)Hh * Hh), blk256, 0, stream>>>(wv, wvt, Hh, Hh);
  wtrans_kernel<<<gblocks((size_t)Hh * Hh), blk256, 0, stream>>>(wo, wot, Hh, Hh);
  wtrans_kernel<<<gblocks((size_t)Hh * FFf), blk256, 0, stream>>>(fw1, fw1t, Hh, FFf);
  wtrans_kernel<<<gblocks((size_t)FFf * Hh), blk256, 0, stream>>>(fw2, fw2t, FFf, Hh);
  wtrans_kernel<<<gblocks((size_t)Hh * Hh), blk256, 0, stream>>>(gaw, gawt, Hh, Hh);
  wtrans_kernel<<<gblocks((size_t)Hh * Hh), blk256, 0, stream>>>(gtw, gtwt, Hh, Hh);

  for (int s = 0; s < NSPANn; ++s) {
    h16* aih = bufA;
    h16* h1h = bufB;
    h16* afh = bufC;
    // audio MLP chain
    cvt_audio_kernel<<<gblocks((size_t)MA * Aa), blk256, 0, stream>>>(audio, aih, s);
    gemm_kernel<1, true, false><<<ggemm(MA, Hh), blk128, 0, stream>>>(
        aih, mw1t, mb1, h1h, nullptr, MA, Hh, Aa);
    gemm_kernel<0, true, false><<<ggemm(MA, Hh), blk128, 0, stream>>>(
        h1h, mw2t, mb2, afh, nullptr, MA, Hh, Hh);
    // K, V projections (reuse dead buffers)
    h16* khb = bufA;
    h16* vhb = bufB;
    gemm_kernel<0, true, false><<<ggemm(MA, Hh), blk128, 0, stream>>>(
        afh, wkt, bk, khb, nullptr, MA, Hh, Hh);
    gemm_kernel<0, true, false><<<ggemm(MA, Hh), blk128, 0, stream>>>(
        afh, wvt, bv, vhb, nullptr, MA, Hh, Hh);
    h16* vth = bufC;  // afh dead now
    vtrans_kernel<<<gblocks((size_t)Bb * NHh * DHh * TAa), blk256, 0, stream>>>(vhb, vth);
    // gather span from evolving hs
    gather_kernel<<<dim3(ML), blk256, 0, stream>>>(hs, spans, s, spanf, spanh);
    // Q projection + attention
    gemm_kernel<0, true, false><<<ggemm(ML, Hh), blk128, 0, stream>>>(
        spanh, wqt, bq, qhb, nullptr, ML, Hh, Hh);
    attn_kernel<<<dim3(Bb * NHh), blk128, 0, stream>>>(qhb, khb, vth, ctxh);
    // output projection, LN1
    gemm_kernel<0, false, true><<<ggemm(ML, Hh), blk128, 0, stream>>>(
        ctxh, wot, bo, nullptr, of, ML, Hh, Hh);
    addln_kernel<<<dim3(ML), blk256, 0, stream>>>(spanf, of, ln1g, ln1b, out1f, out1h);
    // FFN + LN2
    gemm_kernel<2, true, false><<<ggemm(ML, FFf), blk128, 0, stream>>>(
        out1h, fw1t, fb1, ffh, nullptr, ML, FFf, Hh);
    gemm_kernel<0, false, true><<<ggemm(ML, Hh), blk128, 0, stream>>>(
        ffh, fw2t, fb2, nullptr, ff2f, ML, Hh, FFf);
    addln_kernel<<<dim3(ML), blk256, 0, stream>>>(out1f, ff2f, ln2g, ln2b, out2f, out2h);
    // gates
    gemm_kernel<0, false, true><<<ggemm(ML, Hh), blk128, 0, stream>>>(
        out2h, gawt, gab, nullptr, g1f, ML, Hh, Hh);
    gemm_kernel<0, false, true><<<ggemm(ML, Hh), blk128, 0, stream>>>(
        spanh, gtwt, gtb, nullptr, g2f, ML, Hh, Hh);
    // gated scatter-add back into hs
    scatter_kernel<<<dim3(ML), blk256, 0, stream>>>(g1f, g2f, out2f, spanf, spans,
                                                    act, s, hs);
  }
  (void)in_sizes; (void)n_in; (void)out_size; (void)ws_size;
}
